// HydrAMPGRU_35880156791075
// MI455X (gfx1250) — matrix operand order repaired and checked by reading, not compiled, not run on hardware
//
#include <hip/hip_runtime.h>

typedef __attribute__((ext_vector_type(16))) _Float16 v16h;
typedef __attribute__((ext_vector_type(8)))  _Float16 v8h;
typedef __attribute__((ext_vector_type(8)))  float    v8f;

#define UNITS   66
#define U3      198
#define T_STEPS 25
#define XST     104      // half-stride of activation LDS rows (bank-conflict free)
#define OST     84       // float-stride of output staging rows
#define NFRAGS  90       // 6 blocks * 5 unit tiles * 3 K chunks
#define FRAG_BYTES 1024
#define WS_BIAS_OFF ((size_t)NFRAGS * FRAG_BYTES)
#define COPY_SLOTS 7     // ceil(16*66 / 160)

// ---------------------------------------------------------------------------
// Prep: swizzle fp32 weights into CDNA5 f16 WMMA A-fragment layout in d_ws.
// A-matrix 16x32 f16 layout used here (contiguous k, not the ISA table order; B matches):
//   lane<16:  M=lane,    V0..7 -> K=0..15 (pairs)
//   lane>=16: M=lane-16, V0..7 -> K=16..31
// Stored per-lane contiguous: dword = frag*256 + lane*8 + v.
// Blocks: 0=Wz 1=Wr 2=Wh (kernel), 3=Rz 4=Rr 5=Rh (recurrent_kernel).
// A = W^T tile: A[m][k] = W[k][tile*16 + m], zero-padded past 66.
// ---------------------------------------------------------------------------
__global__ void gru_prep_kernel(const float* __restrict__ kern,
                                const float* __restrict__ rk,
                                const float* __restrict__ bias,
                                _Float16* __restrict__ wsf,
                                float* __restrict__ bpad)
{
  int idx = blockIdx.x * blockDim.x + threadIdx.x;
  if (idx < NFRAGS * 256) {
    int fid  = idx >> 8;
    int r    = idx & 255;
    int lane = r >> 3;
    int v    = r & 7;
    int c    = fid % 3;
    int ti   = (fid / 3) % 5;
    int blk  = fid / 15;
    int m    = lane & 15;
    int kb   = (lane >> 4) * 16;
    int koff = kb + 2 * v;
    int k0   = c * 32 + koff;
    int u    = ti * 16 + m;
    const float* src = (blk < 3) ? kern : rk;
    int col0 = (blk % 3) * UNITS;
    float h0 = (k0     < UNITS && u < UNITS) ? src[(size_t)k0 * U3 + col0 + u] : 0.f;
    float h1 = (k0 + 1 < UNITS && u < UNITS) ? src[(size_t)(k0 + 1) * U3 + col0 + u] : 0.f;
    wsf[(size_t)idx * 2]     = (_Float16)h0;
    wsf[(size_t)idx * 2 + 1] = (_Float16)h1;
  } else if (idx < NFRAGS * 256 + 240) {
    int j = idx - NFRAGS * 256;
    int gate = j / 80, u = j % 80;
    bpad[j] = (u < UNITS) ? bias[gate * UNITS + u] : 0.f;
  }
}

__device__ __forceinline__ float sigmoidf_(float x) { return 1.f / (1.f + __expf(-x)); }
__device__ __forceinline__ float tanhf_(float x)    { return 1.f - 2.f / (__expf(2.f * x) + 1.f); }

// ---------------------------------------------------------------------------
// Main fused GRU. 320 threads = 10 waves = 2 groups of 5 waves.
// Group g owns batch rows [Mbase, Mbase+16); wave i in group owns unit tile i.
// Computes D^T tiles: D[m=unit][n=batch] via v_wmma_f32_16x16x32_f16 with the
// weight as A (register-resident) and activations as B (from LDS).
// ---------------------------------------------------------------------------
__global__ __launch_bounds__(320, 1) void gru_main_kernel(
    const float* __restrict__ input_,
    const float* __restrict__ state,
    const _Float16* __restrict__ wsf,
    const float* __restrict__ bpad,
    float* __restrict__ out)
{
  __shared__ _Float16 xbuf[2][16 * XST];
  __shared__ _Float16 hbuf[2][16 * XST];
  __shared__ _Float16 rhbuf[2][16 * XST];
  __shared__ float    obuf[2][16 * OST];

  const int tid  = threadIdx.x;
  const int wave = tid >> 5;
  const int lane = tid & 31;
  const int g    = wave / 5;
  const int i    = wave % 5;          // unit tile 0..4
  const int gtid = tid - g * 160;
  const int Mbase = blockIdx.x * 32 + g * 16;
  const int hi   = lane >> 4;
  const int n    = lane & 15;         // batch column within tile (D layout)

  // zero-init activation buffers (padding columns must stay 0 forever)
  for (int idx = gtid; idx < 16 * XST; idx += 160) {
    xbuf[g][idx]  = (_Float16)0.f;
    hbuf[g][idx]  = (_Float16)0.f;
    rhbuf[g][idx] = (_Float16)0.f;
  }
  __syncthreads();
  // step 0 operands: x = input_, h = state  (state==0 => recurrent terms vanish)
  for (int idx = gtid; idx < 16 * UNITS; idx += 160) {
    int b = idx / UNITS, u = idx % UNITS;
    xbuf[g][b * XST + u] = (_Float16)input_[(size_t)(Mbase + b) * UNITS + u];
    hbuf[g][b * XST + u] = (_Float16)state [(size_t)(Mbase + b) * UNITS + u];
  }

  // persistent weight A-fragments: 6 blocks x 3 K-chunks = 144 VGPRs
  v16h af[6][3];
  #pragma unroll
  for (int blk = 0; blk < 6; ++blk)
    #pragma unroll
    for (int c = 0; c < 3; ++c)
      af[blk][c] = *(const v16h*)((const char*)wsf +
                    (size_t)((blk * 5 + i) * 3 + c) * FRAG_BYTES + lane * 32);

  // biases in D layout (row m = e + 8*hi within tile i)
  float bz[8], br[8], bh[8];
  #pragma unroll
  for (int e = 0; e < 8; ++e) {
    int u = i * 16 + e + hi * 8;
    bz[e] = bpad[u];
    br[e] = bpad[80 + u];
    bh[e] = bpad[160 + u];
  }

  // h tile in D-layout f32 registers, init from state
  v8f h;
  #pragma unroll
  for (int e = 0; e < 8; ++e) {
    int u = i * 16 + e + hi * 8;
    h[e] = (u < UNITS) ? state[(size_t)(Mbase + n) * UNITS + u] : 0.f;
  }

  // precompute per-thread output-copy slots (fixed (b,u) pairs; only t varies)
  int   cp_oidx[COPY_SLOTS];   // obuf index
  size_t cp_obase[COPY_SLOTS]; // out base element offset (add t*UNITS)
  bool  cp_on[COPY_SLOTS];
  #pragma unroll
  for (int s = 0; s < COPY_SLOTS; ++s) {
    int idx = gtid + s * 160;
    cp_on[s] = (idx < 16 * UNITS);
    int b = cp_on[s] ? (idx / UNITS) : 0;
    int u = cp_on[s] ? (idx % UNITS) : 0;
    cp_oidx[s]  = b * OST + u;
    cp_obase[s] = (size_t)(Mbase + b) * T_STEPS * UNITS + u;
  }

  for (int t = 0; t < T_STEPS; ++t) {
    __syncthreads();  // x/h buffers ready; previous obuf copy done

    // B fragments: lane holds batch col n, K = 16*hi + [0..15] contiguous
    v16h bx[3], bhv[3];
    #pragma unroll
    for (int c = 0; c < 3; ++c) {
      const _Float16* px = &xbuf[g][n * XST + c * 32 + hi * 16];
      v8h xlo = *(const v8h*)px;
      v8h xhi = *(const v8h*)(px + 8);
      bx[c] = __builtin_shufflevector(xlo, xhi, 0,1,2,3,4,5,6,7,8,9,10,11,12,13,14,15);
      const _Float16* ph = &hbuf[g][n * XST + c * 32 + hi * 16];
      v8h hlo = *(const v8h*)ph;
      v8h hhi = *(const v8h*)(ph + 8);
      bhv[c] = __builtin_shufflevector(hlo, hhi, 0,1,2,3,4,5,6,7,8,9,10,11,12,13,14,15);
    }

    // fused gate GEMMs: z and r accumulate both kernel and recurrent products
    v8f zacc = {}, racc = {}, hhacc = {};
    #pragma unroll
    for (int c = 0; c < 3; ++c) {
      zacc  = __builtin_amdgcn_wmma_f32_16x16x32_f16(false, af[0][c], false, bx[c],  (short)0, zacc,  false, false);
      zacc  = __builtin_amdgcn_wmma_f32_16x16x32_f16(false, af[3][c], false, bhv[c], (short)0, zacc,  false, false);
      racc  = __builtin_amdgcn_wmma_f32_16x16x32_f16(false, af[1][c], false, bx[c],  (short)0, racc,  false, false);
      racc  = __builtin_amdgcn_wmma_f32_16x16x32_f16(false, af[4][c], false, bhv[c], (short)0, racc,  false, false);
      hhacc = __builtin_amdgcn_wmma_f32_16x16x32_f16(false, af[2][c], false, bx[c],  (short)0, hhacc, false, false);
    }

    v8f z, rr;
    v8h rh16;
    #pragma unroll
    for (int e = 0; e < 8; ++e) {
      z[e]  = sigmoidf_(zacc[e] + bz[e]);
      rr[e] = sigmoidf_(racc[e] + br[e]);
      rh16[e] = (_Float16)(rr[e] * h[e]);
    }
    // D-layout row m = e+8*hi maps to units u = i*16+8*hi+e -> contiguous 16B store
    *(v8h*)(&rhbuf[g][n * XST + i * 16 + hi * 8]) = rh16;
    __syncthreads();  // rhbuf ready across waves

    #pragma unroll
    for (int c = 0; c < 3; ++c) {
      const _Float16* pr = &rhbuf[g][n * XST + c * 32 + hi * 16];
      v8h rlo = *(const v8h*)pr;
      v8h rhi = *(const v8h*)(pr + 8);
      v16h brh = __builtin_shufflevector(rlo, rhi, 0,1,2,3,4,5,6,7,8,9,10,11,12,13,14,15);
      hhacc = __builtin_amdgcn_wmma_f32_16x16x32_f16(false, af[5][c], false, brh, (short)0, hhacc, false, false);
    }

    v8h hn16;
    #pragma unroll
    for (int e = 0; e < 8; ++e) {
      float hhv = tanhf_(hhacc[e] + bh[e]);
      float hn  = z[e] * h[e] + (1.f - z[e]) * hhv;
      h[e] = hn;
      hn16[e] = (_Float16)hn;
      obuf[g][n * OST + i * 16 + hi * 8 + e] = hn;
    }
    // next step: x == h
    *(v8h*)(&xbuf[g][n * XST + i * 16 + hi * 8]) = hn16;
    *(v8h*)(&hbuf[g][n * XST + i * 16 + hi * 8]) = hn16;
    __syncthreads();  // obuf ready

    // coalesced output: out[b, t, u] with precomputed slots
    const size_t tofs = (size_t)t * UNITS;
    #pragma unroll
    for (int s = 0; s < COPY_SLOTS; ++s)
      if (cp_on[s])
        out[cp_obase[s] + tofs] = obuf[g][cp_oidx[s]];
  }
}

extern "C" void kernel_launch(void* const* d_in, const int* in_sizes, int n_in,
                              void* d_out, int out_size, void* d_ws, size_t ws_size,
                              hipStream_t stream) {
  (void)in_sizes; (void)n_in; (void)out_size; (void)ws_size;
  const float* input_ = (const float*)d_in[0];
  const float* state  = (const float*)d_in[1];
  const float* kern   = (const float*)d_in[2];
  const float* rk     = (const float*)d_in[3];
  const float* bias   = (const float*)d_in[4];
  _Float16* wsf  = (_Float16*)d_ws;
  float*    bpad = (float*)((char*)d_ws + WS_BIAS_OFF);
  float*    out  = (float*)d_out;

  int prep_threads = NFRAGS * 256 + 240;
  gru_prep_kernel<<<(prep_threads + 255) / 256, 256, 0, stream>>>(kern, rk, bias, wsf, bpad);
  // 131072 batch rows / (2 groups * 16 rows) = 4096 workgroups
  gru_main_kernel<<<4096, 320, 0, stream>>>(input_, state, wsf, bpad, out);
}
